// GAT_25220047962614
// MI455X (gfx1250) — matrix operand order repaired and checked by reading, not compiled, not run on hardware
//
#include <hip/hip_runtime.h>
#include <hip/hip_bf16.h>
#include <climits>

// ---------------------------------------------------------------------------
// GAT (2-layer) for MI455X / gfx1250.
// GEMMs: bf16 WMMA (v_wmma_f32_16x16x32_bf16), fp32 accumulate.
//   - A pre-converted to bf16 row-major (2x global_load_b128 per lane/k-step)
//   - B pre-packed into WMMA fragment order (2x global_load_b128, no masking)
//   - NT output tiles per wave to reuse the A fragment across WMMAs
// Softmax-scatter: ordered-int atomicMax + fp32 atomicAdd (L2-resident).
// ---------------------------------------------------------------------------

typedef __attribute__((ext_vector_type(16))) __bf16 v16bf;
typedef __attribute__((ext_vector_type(8)))  __bf16 bf16x8;
typedef __attribute__((ext_vector_type(8)))  float  v8f;

#define NEG_SLOPE 0.2f

// ----------------------- small utility kernels -----------------------------

__global__ void fill_f32(float* __restrict__ p, float v, int n) {
    int i = blockIdx.x * blockDim.x + threadIdx.x;
    if (i < n) p[i] = v;
}

__global__ void fill_i32(int* __restrict__ p, int v, int n) {
    int i = blockIdx.x * blockDim.x + threadIdx.x;
    if (i < n) p[i] = v;
}

// fp32 -> bf16, elementwise (A-operand pre-conversion)
__global__ void cvt_bf16(const float* __restrict__ in, __bf16* __restrict__ out, int n) {
    int i = blockIdx.x * blockDim.x + threadIdx.x;
    if (i < n) out[i] = (__bf16)in[i];
}

// Pack fp32 B[K,N] into WMMA B-fragment order, bf16, zero-padded.
// Layout: [tk][tn][lane][i] ; contiguous k order, same as the A loads (B 32x16 bf16):
//   col = tn*16 + (lane&15) ; K = tk*32 + 16*(lane>=16) + i
__global__ void pack_B(const float* __restrict__ B, __bf16* __restrict__ Bpk,
                       int K, int N, int tilesN, int total) {
    int t = blockIdx.x * blockDim.x + threadIdx.x;
    if (t >= total) return;
    int frag = t >> 9;          // which 16x32 fragment (512 bf16 each)
    int r    = t & 511;
    int lane = r >> 4;
    int i    = r & 15;
    int tk = frag / tilesN;
    int tn = frag % tilesN;
    int k  = tk * 32 + ((lane >= 16) ? 16 : 0) + i;
    int n  = tn * 16 + (lane & 15);
    Bpk[t] = (k < K && n < N) ? (__bf16)B[(size_t)k * N + n] : (__bf16)0.0f;
}

// Monotone order-preserving float<->int key for atomicMax on floats.
__device__ inline int f32_key(float f) {
    int i = __float_as_int(f);
    return (i >= 0) ? i : (i ^ 0x7fffffff);
}
__device__ inline float key_f32(int k) {
    return __int_as_float((k >= 0) ? k : (k ^ 0x7fffffff));
}

// ----------------------- WMMA bf16 GEMM ------------------------------------
// C[M,N] = A[M,K] @ B[K,N]. A: bf16 row-major. Bpk: fragment-packed bf16.
// One wave32 computes NT adjacent 16x16 tiles (one A fragment reused).
// M, K multiples of 16/32; N handled via store guard (Bpk is zero-padded).
template <int NT>
__global__ __launch_bounds__(128) void wmma_gemm_pk(
    const __bf16* __restrict__ A, const __bf16* __restrict__ Bpk,
    float* __restrict__ C, int M, int N, int K,
    int tilesN, int totalWaves)
{
    int wave = blockIdx.x * (blockDim.x >> 5) + (threadIdx.x >> 5);
    if (wave >= totalWaves) return;            // wave-uniform exit

    int groupsN = tilesN / NT;
    int tm = wave / groupsN;
    int tg = wave % groupsN;
    int lane = threadIdx.x & 31;
    int hs   = lane >> 4;
    int l15  = lane & 15;

    v8f acc[NT];
#pragma unroll
    for (int j = 0; j < NT; ++j) acc[j] = (v8f){};

    // A fragment pointer: row-major bf16 row, in bf16x8 (16-byte) units.
    const bf16x8* Ab = (const bf16x8*)(A + (size_t)(tm * 16 + l15) * K);
    const bf16x8* Bb = (const bf16x8*)Bpk;

    int tilesK = K >> 5;
#pragma unroll
    for (int tk = 0; tk < 4; ++tk) {           // K=128 always here
        if (tk >= tilesK) break;
        __builtin_prefetch((const void*)(Ab + (tk + 1) * 4), 0, 1);
        // A 16x32 bf16 fragment: K {hs*16 .. +15} (contiguous, matches Bpk)
        bf16x8 alo = Ab[tk * 4 + 2 * hs];
        bf16x8 ahi = Ab[tk * 4 + 2 * hs + 1];
        v16bf a = __builtin_shufflevector(alo, ahi,
                    0,1,2,3,4,5,6,7,8,9,10,11,12,13,14,15);
#pragma unroll
        for (int j = 0; j < NT; ++j) {
            const bf16x8* bp = Bb + ((size_t)(tk * tilesN + tg * NT + j) * 32 + lane) * 2;
            v16bf b = __builtin_shufflevector(bp[0], bp[1],
                        0,1,2,3,4,5,6,7,8,9,10,11,12,13,14,15);
            acc[j] = __builtin_amdgcn_wmma_f32_16x16x32_bf16(
                false, a, false, b, (short)0, acc[j], false, false);
        }
    }

#pragma unroll
    for (int j = 0; j < NT; ++j) {
        int col = (tg * NT + j) * 16 + l15;
        if (col < N) {
#pragma unroll
            for (int r = 0; r < 8; ++r) {
                int row = tm * 16 + hs * 8 + r;
                C[(size_t)row * N + col] = acc[j][r];
            }
        }
    }
}

// ----------------------- layer 1: attention scores -------------------------
__global__ void att_scores_l1(const float* __restrict__ h,
                              const float* __restrict__ att_s,
                              const float* __restrict__ att_d,
                              float* __restrict__ a_s, float* __restrict__ a_d,
                              int Nn)
{
    int t = blockIdx.x * blockDim.x + threadIdx.x;
    if (t >= Nn * 8) return;
    int n = t >> 3, hh = t & 7;
    const float* hp = h + (size_t)n * 128 + hh * 16;
    const float* as = att_s + hh * 16;
    const float* ad = att_d + hh * 16;
    float ss = 0.f, dd = 0.f;
#pragma unroll
    for (int c = 0; c < 16; ++c) {
        float v = hp[c];
        ss += v * as[c];
        dd += v * ad[c];
    }
    a_s[t] = ss;
    a_d[t] = dd;
}

// ----------------------- layer 1: edge passes ------------------------------

__device__ inline float leaky(float x) { return x > 0.f ? x : NEG_SLOPE * x; }

__global__ void edge1_max(const int* __restrict__ src, const int* __restrict__ dst,
                          const float* __restrict__ a_s, const float* __restrict__ a_d,
                          int* __restrict__ nmax, int E)
{
    int t = blockIdx.x * blockDim.x + threadIdx.x;
    if (t >= E * 8) return;
    int e = t >> 3, h = t & 7;
    int d = dst[e];
    float l = leaky(a_s[src[e] * 8 + h] + a_d[d * 8 + h]);
    atomicMax(&nmax[d * 8 + h], f32_key(l));
}

__global__ void edge1_sum(const int* __restrict__ src, const int* __restrict__ dst,
                          const float* __restrict__ a_s, const float* __restrict__ a_d,
                          const int* __restrict__ nmax, float* __restrict__ nsum, int E)
{
    int t = blockIdx.x * blockDim.x + threadIdx.x;
    if (t >= E * 8) return;
    int e = t >> 3, h = t & 7;
    int d = dst[e];
    float l = leaky(a_s[src[e] * 8 + h] + a_d[d * 8 + h]);
    atomicAdd(&nsum[d * 8 + h], __expf(l - key_f32(nmax[d * 8 + h])));
}

__global__ void edge1_agg(const int* __restrict__ src, const int* __restrict__ dst,
                          const float* __restrict__ a_s, const float* __restrict__ a_d,
                          const int* __restrict__ nmax, const float* __restrict__ nsum,
                          const float* __restrict__ hlin, float* __restrict__ out1, int E)
{
    int t = blockIdx.x * blockDim.x + threadIdx.x;
    if (t >= E * 8) return;
    int e = t >> 3, h = t & 7;
    int s = src[e], d = dst[e];
    float l = leaky(a_s[s * 8 + h] + a_d[d * 8 + h]);
    float w = __expf(l - key_f32(nmax[d * 8 + h])) / nsum[d * 8 + h];
    const float* hp = hlin + (size_t)s * 128 + h * 16;
    float* op = out1 + (size_t)d * 128 + h * 16;
#pragma unroll
    for (int c = 0; c < 16; ++c)
        atomicAdd(&op[c], w * hp[c]);
}

// out1 = elu(out1 + b1)
__global__ void finalize1(float* __restrict__ out1, const float* __restrict__ b1, int Nn)
{
    int t = blockIdx.x * blockDim.x + threadIdx.x;
    if (t >= Nn * 128) return;
    float v = out1[t] + b1[t & 127];
    out1[t] = v > 0.f ? v : expm1f(v);
}

// ----------------------- layer 2 (H=1, C=40) -------------------------------

__global__ void att_scores_l2(const float* __restrict__ t2,
                              const float* __restrict__ att_s,
                              const float* __restrict__ att_d,
                              float* __restrict__ a_s, float* __restrict__ a_d,
                              int Nn)
{
    int n = blockIdx.x * blockDim.x + threadIdx.x;
    if (n >= Nn) return;
    const float* tp = t2 + (size_t)n * 40;
    float ss = 0.f, dd = 0.f;
#pragma unroll
    for (int c = 0; c < 40; ++c) {
        float v = tp[c];
        ss += v * att_s[c];
        dd += v * att_d[c];
    }
    a_s[n] = ss;
    a_d[n] = dd;
}

__global__ void edge2_max(const int* __restrict__ src, const int* __restrict__ dst,
                          const float* __restrict__ a_s, const float* __restrict__ a_d,
                          int* __restrict__ nmax, int E)
{
    int e = blockIdx.x * blockDim.x + threadIdx.x;
    if (e >= E) return;
    int d = dst[e];
    float l = leaky(a_s[src[e]] + a_d[d]);
    atomicMax(&nmax[d], f32_key(l));
}

__global__ void edge2_sum(const int* __restrict__ src, const int* __restrict__ dst,
                          const float* __restrict__ a_s, const float* __restrict__ a_d,
                          const int* __restrict__ nmax, float* __restrict__ nsum, int E)
{
    int e = blockIdx.x * blockDim.x + threadIdx.x;
    if (e >= E) return;
    int d = dst[e];
    float l = leaky(a_s[src[e]] + a_d[d]);
    atomicAdd(&nsum[d], __expf(l - key_f32(nmax[d])));
}

__global__ void edge2_agg(const int* __restrict__ src, const int* __restrict__ dst,
                          const float* __restrict__ a_s, const float* __restrict__ a_d,
                          const int* __restrict__ nmax, const float* __restrict__ nsum,
                          const float* __restrict__ t2, float* __restrict__ out, int E)
{
    int e = blockIdx.x * blockDim.x + threadIdx.x;
    if (e >= E) return;
    int s = src[e], d = dst[e];
    float l = leaky(a_s[s] + a_d[d]);
    float w = __expf(l - key_f32(nmax[d])) / nsum[d];
    const float* tp = t2 + (size_t)s * 40;
    float* op = out + (size_t)d * 40;
#pragma unroll
    for (int c = 0; c < 40; ++c)
        atomicAdd(&op[c], w * tp[c]);
}

__global__ void finalize2(float* __restrict__ out, const float* __restrict__ b2, int Nn)
{
    int t = blockIdx.x * blockDim.x + threadIdx.x;
    if (t >= Nn * 40) return;
    out[t] += b2[t % 40];
}

// ---------------------------------------------------------------------------

extern "C" void kernel_launch(void* const* d_in, const int* in_sizes, int n_in,
                              void* d_out, int out_size, void* d_ws, size_t ws_size,
                              hipStream_t stream)
{
    const float* x   = (const float*)d_in[0];
    const int*   ei  = (const int*)d_in[1];
    const float* W1  = (const float*)d_in[2];
    const float* as1 = (const float*)d_in[3];
    const float* ad1 = (const float*)d_in[4];
    const float* b1  = (const float*)d_in[5];
    const float* W2  = (const float*)d_in[6];
    const float* as2 = (const float*)d_in[7];
    const float* ad2 = (const float*)d_in[8];
    const float* b2  = (const float*)d_in[9];

    const int Nn = in_sizes[0] / 128;   // 50000
    const int E  = in_sizes[1] / 2;     // 800000
    const int* src = ei;
    const int* dst = ei + E;

    // ---- workspace carving (256B aligned) ----
    char* w = (char*)d_ws;
    auto carve = [&](size_t bytes) -> char* {
        char* p = w;
        w += (bytes + 255) & ~(size_t)255;
        return p;
    };
    float*  hlin = (float*)carve((size_t)Nn * 128 * 4);   // layer1 linear feats (reused as t2)
    float*  out1 = (float*)carve((size_t)Nn * 128 * 4);   // layer1 aggregated -> elu
    __bf16* Abf  = (__bf16*)carve((size_t)Nn * 128 * 2);  // bf16 A operand (x, then elu(out1))
    __bf16* B1pk = (__bf16*)carve((size_t)4 * 8 * 512 * 2);
    __bf16* B2pk = (__bf16*)carve((size_t)4 * 3 * 512 * 2);
    float*  aS1  = (float*)carve((size_t)Nn * 8 * 4);
    float*  aD1  = (float*)carve((size_t)Nn * 8 * 4);
    int*    mx1  = (int*)  carve((size_t)Nn * 8 * 4);
    float*  sm1  = (float*)carve((size_t)Nn * 8 * 4);
    float*  aS2  = (float*)carve((size_t)Nn * 4);
    float*  aD2  = (float*)carve((size_t)Nn * 4);
    int*    mx2  = (int*)  carve((size_t)Nn * 4);
    float*  sm2  = (float*)carve((size_t)Nn * 4);
    float*  t2   = hlin;                                  // dead after edge1_agg/finalize1
    float*  out  = (float*)d_out;

    const int TB = 256;
    auto blocks = [&](long long n) { return (int)((n + TB - 1) / TB); };

    // ---- zero / init accumulators (every call: graph-replay deterministic) ----
    fill_f32<<<blocks((long long)Nn * 128), TB, 0, stream>>>(out1, 0.f, Nn * 128);
    fill_f32<<<blocks(Nn * 8), TB, 0, stream>>>(sm1, 0.f, Nn * 8);
    fill_i32<<<blocks(Nn * 8), TB, 0, stream>>>(mx1, INT_MIN, Nn * 8);
    fill_f32<<<blocks(Nn), TB, 0, stream>>>(sm2, 0.f, Nn);
    fill_i32<<<blocks(Nn), TB, 0, stream>>>(mx2, INT_MIN, Nn);
    fill_f32<<<blocks((long long)Nn * 40), TB, 0, stream>>>(out, 0.f, Nn * 40);

    // ---- layer 1 ----
    cvt_bf16<<<blocks((long long)Nn * 128), TB, 0, stream>>>(x, Abf, Nn * 128);
    {
        int totB = 4 * 8 * 512;
        pack_B<<<blocks(totB), TB, 0, stream>>>(W1, B1pk, 128, 128, 8, totB);
        int tilesM = Nn / 16, tilesN = 8, NT = 4;
        int waves = tilesM * (tilesN / NT);              // 6250
        wmma_gemm_pk<4><<<(waves + 3) / 4, 128, 0, stream>>>(
            Abf, B1pk, hlin, Nn, 128, 128, tilesN, waves);
    }
    att_scores_l1<<<blocks(Nn * 8), TB, 0, stream>>>(hlin, as1, ad1, aS1, aD1, Nn);
    edge1_max<<<blocks((long long)E * 8), TB, 0, stream>>>(src, dst, aS1, aD1, mx1, E);
    edge1_sum<<<blocks((long long)E * 8), TB, 0, stream>>>(src, dst, aS1, aD1, mx1, sm1, E);
    edge1_agg<<<blocks((long long)E * 8), TB, 0, stream>>>(src, dst, aS1, aD1, mx1, sm1,
                                                           hlin, out1, E);
    finalize1<<<blocks((long long)Nn * 128), TB, 0, stream>>>(out1, b1, Nn);

    // ---- layer 2 ----
    cvt_bf16<<<blocks((long long)Nn * 128), TB, 0, stream>>>(out1, Abf, Nn * 128);
    {
        int totB = 4 * 3 * 512;
        pack_B<<<blocks(totB), TB, 0, stream>>>(W2, B2pk, 128, 40, 3, totB);
        int tilesM = Nn / 16, tilesN = 3, NT = 3;
        int waves = tilesM * (tilesN / NT);              // 3125
        wmma_gemm_pk<3><<<(waves + 3) / 4, 128, 0, stream>>>(
            Abf, B2pk, t2, Nn, 40, 128, tilesN, waves);
    }
    att_scores_l2<<<blocks(Nn), TB, 0, stream>>>(t2, as2, ad2, aS2, aD2, Nn);
    edge2_max<<<blocks(E), TB, 0, stream>>>(src, dst, aS2, aD2, mx2, E);
    edge2_sum<<<blocks(E), TB, 0, stream>>>(src, dst, aS2, aD2, mx2, sm2, E);
    edge2_agg<<<blocks(E), TB, 0, stream>>>(src, dst, aS2, aD2, mx2, sm2, t2, out, E);
    finalize2<<<blocks((long long)Nn * 40), TB, 0, stream>>>(out, b2, Nn);

    (void)n_in; (void)out_size; (void)ws_size;
}
